// Block_41197326303280
// MI455X (gfx1250) — matrix operand order repaired and checked by reading, not compiled, not run on hardware
//
#include <hip/hip_runtime.h>

#define B_     2
#define CIN    64
#define COUT   128
#define H_     224
#define W_     224
#define HW     (H_*W_)
#define WS     7
#define KNN    20
#define PAD    3
#define NH     32
#define NW     32
#define NPATCH (NH*NW)
#define PP     49

typedef __attribute__((ext_vector_type(16))) _Float16 v16h;
typedef __attribute__((ext_vector_type(8)))  _Float16 v8h;
typedef __attribute__((ext_vector_type(8)))  float    v8f;
typedef __attribute__((ext_vector_type(4)))  float    v4f;

// ---------------------------------------------------------------------------
// K1: per (b, patch): Gram(49x49 over 64ch) via WMMA f16 -> pairwise dist ->
//     top-20 neighbor indices per pixel.
//     sFt is [p][c] (k-fastest, 72-half row stride: 16B-aligned, conflict-free
//     b128 fragment reads).  pd is symmetric, so the C tile is stored
//     TRANSPOSED into sPD with vector stores (lane's 8 outputs contiguous).
// ---------------------------------------------------------------------------
__global__ __launch_bounds__(128)
void k1_gram_topk(const float* __restrict__ x, int* __restrict__ gidx) {
  __shared__ alignas(16) _Float16 sFt[64][72];  // [p][c], p padded 49->64
  __shared__ alignas(16) float sXX[64];
  __shared__ alignas(16) float sPD[64][68];     // [q][p], padded row

  const int bn = blockIdx.x;
  const int b  = bn / NPATCH;
  const int n  = bn % NPATCH;
  const int th = n / NW, tw = n % NW;
  const int tid = threadIdx.x;

  // stage F^T: consecutive threads write consecutive c (merged LDS writes)
  for (int i = tid; i < 64*64; i += 128) {
    const int p = i >> 6, c = i & 63;
    float v = 0.f;
    if (p < PP) {
      const int y  = th*WS + p/WS - PAD;
      const int xw = tw*WS + p%WS - PAD;
      if (y >= 0 && y < H_ && xw >= 0 && xw < W_)
        v = x[((b*CIN + c)*H_ + y)*W_ + xw];
    }
    sFt[p][c] = (_Float16)v;
  }
  __syncthreads();

  if (tid < 64) {                       // ||f_p||^2 ; contiguous LDS reads
    float s = 0.f;
    for (int c = 0; c < 64; ++c) { const float f = (float)sFt[tid][c]; s += f*f; }
    sXX[tid] = s;
  }
  __syncthreads();

  const int wv = tid >> 5;              // 4 waves: wave = M-tile
  const int lane = tid & 31;
  const int hi = lane >> 4;
  const int lm = lane & 15;

  v8f zero = {};
  v8f acc[4] = {zero, zero, zero, zero};

  const _Float16* rowA = sFt[wv*16 + lm];
  #pragma unroll
  for (int ks = 0; ks < 2; ++ks) {      // K = 64 in 2 steps of 32
    const int cb = ks*32;
    // A fragment: two aligned v8h chunks (K = hi*16+0..7 and hi*16+8..15)
    const v8h alo = *(const v8h*)(rowA + cb + hi*16);
    const v8h ahi = *(const v8h*)(rowA + cb + hi*16 + 8);
    const v16h a = __builtin_shufflevector(alo, ahi,
        0,1,2,3,4,5,6,7,8,9,10,11,12,13,14,15);
    #pragma unroll
    for (int nt = 0; nt < 4; ++nt) {
      // B fragment: K = hi*16+e contiguous -> one aligned v16h (2x b128)
      const v16h bf = *(const v16h*)(sFt[nt*16 + lm] + cb + hi*16);
      acc[nt] = __builtin_amdgcn_wmma_f32_16x16x32_f16(
          false, a, false, bf, (short)0, acc[nt], false, false);
    }
  }

  // pd[p][q] = 2G[pq] - xx[p] - xx[q]; symmetric -> store transposed,
  // vectorized, branch-free (padded region is never read by top-k).
  const int mrow = wv*16 + hi*8;        // this lane's 8 contiguous p values
  const v4f xlo = *(const v4f*)&sXX[mrow];
  const v4f xhi = *(const v4f*)&sXX[mrow + 4];
  #pragma unroll
  for (int nt = 0; nt < 4; ++nt) {
    const int qc = nt*16 + lm;
    const float xq = sXX[qc];
    v4f lo, hi4;
    #pragma unroll
    for (int v = 0; v < 4; ++v) {
      lo[v]  = 2.0f*acc[nt][v]   - xlo[v] - xq;
      hi4[v] = 2.0f*acc[nt][v+4] - xhi[v] - xq;
    }
    *(v4f*)&sPD[qc][mrow]     = lo;
    *(v4f*)&sPD[qc][mrow + 4] = hi4;
  }
  __syncthreads();

  if (tid < PP) {                       // top-20 per row, stable ties
    const int r = tid;
    const int base = (bn*PP + r)*KNN;
    for (int kk = 0; kk < KNN; ++kk) {
      float best = -3.4e38f; int bi = 0;
      for (int q = 0; q < PP; ++q) {
        const float v = sPD[r][q];
        if (v > best) { best = v; bi = q; }
      }
      gidx[base + kk] = bi;
      sPD[r][bi] = -3.4e38f;
    }
  }
}

// ---------------------------------------------------------------------------
// K2: gather neighbors, channel-L2 normalize (wave32 shfl reduce),
//     20x20 FC + relu + max over j  ->  z1 (B,CIN,H,W) stored as f16
// ---------------------------------------------------------------------------
__global__ __launch_bounds__(256)
void k2_gather_fc(const float* __restrict__ x, const int* __restrict__ gidx,
                  const float* __restrict__ fcw, const float* __restrict__ fcb,
                  _Float16* __restrict__ z1h) {
  __shared__ float sW[KNN*KNN];
  __shared__ float sFB[KNN];
  const int tid = threadIdx.x;
  if (tid < KNN*KNN) sW[tid] = fcw[tid];
  if (tid < KNN)     sFB[tid] = fcb[tid];
  __syncthreads();

  const int wv   = tid >> 5;
  const int lane = tid & 31;
  const int pix  = blockIdx.x * 8 + wv;     // one wave per output pixel
  const int b  = pix >= HW ? 1 : 0;
  const int hw = pix - b*HW;
  const int h = hw / W_, w = hw % W_;
  const int hp = h + PAD, wp = w + PAD;
  const bool inc = (hp < NH*WS) && (wp < NW*WS);  // rows/cols >=221 fold to 0

  int th = 0, tw = 0, p = 0;
  if (inc) { th = hp / WS; tw = wp / WS; p = (hp % WS)*WS + (wp % WS); }
  const int ibase = ((b*NPATCH + th*NW + tw)*PP + p)*KNN;
  const int c0 = lane, c1 = lane + 32;

  float g0[KNN], g1[KNN];
  #pragma unroll
  for (int k = 0; k < KNN; ++k) {
    float a = 0.f, bb = 0.f;
    if (inc) {
      const int q  = gidx[ibase + k];
      const int y  = th*WS + q/WS - PAD;
      const int xw = tw*WS + q%WS - PAD;
      if (y >= 0 && y < H_ && xw >= 0 && xw < W_) {
        a  = x[((b*CIN + c0)*H_ + y)*W_ + xw];
        bb = x[((b*CIN + c1)*H_ + y)*W_ + xw];
      }
    }
    g0[k] = a; g1[k] = bb;
  }

  float inv[KNN];
  #pragma unroll
  for (int k = 0; k < KNN; ++k) {           // norm over 64 channels
    float s = g0[k]*g0[k] + g1[k]*g1[k];
    s += __shfl_xor(s, 16, 32);
    s += __shfl_xor(s,  8, 32);
    s += __shfl_xor(s,  4, 32);
    s += __shfl_xor(s,  2, 32);
    s += __shfl_xor(s,  1, 32);
    inv[k] = 1.0f / fmaxf(sqrtf(s), 1e-12f);
  }
  #pragma unroll
  for (int k = 0; k < KNN; ++k) { g0[k] *= inv[k]; g1[k] *= inv[k]; }

  float m0 = -3.4e38f, m1 = -3.4e38f;
  #pragma unroll
  for (int j = 0; j < KNN; ++j) {
    float s0 = sFB[j], s1 = s0;
    #pragma unroll
    for (int k = 0; k < KNN; ++k) {
      const float wjk = sW[j*KNN + k];
      s0 = fmaf(g0[k], wjk, s0);
      s1 = fmaf(g1[k], wjk, s1);
    }
    m0 = fmaxf(m0, s0); m1 = fmaxf(m1, s1);
  }
  z1h[((b*CIN + c0)*H_ + h)*W_ + w] = (_Float16)fmaxf(m0, 0.f);
  z1h[((b*CIN + c1)*H_ + h)*W_ + w] = (_Float16)fmaxf(m1, 0.f);
}

// ---------------------------------------------------------------------------
// K3: 3x3 conv 64->128 as implicit GEMM, WMMA f32_16x16x32_f16.
//     Block tile: M=64 pixels x N=128 couts; K = 576 in 18 steps of 32
//     (fixed (kh,kw) per step).  Double-buffered LDS: stage(s+1) issues
//     right after the barrier, overlapping the WMMAs of step s.
// ---------------------------------------------------------------------------
__global__ __launch_bounds__(256)
void k3_conv_wmma(const _Float16* __restrict__ z1h, const float* __restrict__ cw,
                  const float* __restrict__ cb, float* __restrict__ z2) {
  __shared__ alignas(16) _Float16 sA [2][64][40];    // [buf][pixel][k natural]
  __shared__ alignas(16) _Float16 sBt[2][COUT][40];  // [buf][cout][k natural]

  const int tid  = threadIdx.x;
  const int wv   = tid >> 5;          // wave = N-subtile (cout group)
  const int lane = tid & 31;
  const int hi = lane >> 4;
  const int lm = lane & 15;
  const int mbase = blockIdx.x * 64;

  // per-thread pixel decode for A staging (constant across k-steps)
  const int pixA = tid & 63;
  const int idA  = mbase + pixA;
  const int bA   = idA >= HW ? 1 : 0;
  const int hwA  = idA - bA*HW;
  const int hA   = hwA / W_;
  const int wA   = hwA % W_;

  auto stageA = [&](int s, int buf) {
    const int r   = s >> 1;
    const int cb0 = (s & 1) * 32;
    const int y  = hA + r/3 - 1;
    const int xw = wA + r%3 - 1;
    const bool ok = (y >= 0 && y < H_ && xw >= 0 && xw < W_);
    const int base = ok ? ((bA*CIN)*H_ + y)*W_ + xw : 0;
    for (int cg = tid >> 6; cg < 32; cg += 4) {
      const int pos = cg;
      sA[buf][pixA][pos] = ok ? z1h[base + (cb0 + cg)*HW] : (_Float16)0.f;
    }
  };
  auto stageB = [&](int s, int buf) {
    const int r   = s >> 1;
    const int cb0 = (s & 1) * 32;
    for (int i = tid; i < 32*COUT; i += 256) {
      const int kk = i & 31, co = i >> 5;
      sBt[buf][co][kk] = (_Float16)cw[(co*CIN + cb0 + kk)*9 + r];
    }
  };

  v8f zero = {};
  v8f acc[4] = {zero, zero, zero, zero};

  stageA(0, 0);
  stageB(0, 0);
  for (int s = 0; s < 18; ++s) {
    const int cur = s & 1;
    __syncthreads();                    // stage(s) complete; buf cur^1 free
    if (s + 1 < 18) { stageA(s+1, cur^1); stageB(s+1, cur^1); }

    const v16h bf = *(const v16h*)(sBt[cur][wv*16 + lm] + hi*16);
    #pragma unroll
    for (int mt = 0; mt < 4; ++mt) {
      const v16h af = *(const v16h*)(sA[cur][mt*16 + lm] + hi*16);
      acc[mt] = __builtin_amdgcn_wmma_f32_16x16x32_f16(
          false, af, false, bf, (short)0, acc[mt], false, false);
    }
  }

  const int co = wv*16 + lm;
  const float bias = cb[co];
  #pragma unroll
  for (int mt = 0; mt < 4; ++mt) {
    #pragma unroll
    for (int v = 0; v < 8; ++v) {
      const int id = mbase + mt*16 + hi*8 + v;
      const int b  = id >= HW ? 1 : 0;
      const int hw = id - b*HW;
      z2[(b*COUT + co)*HW + hw] = acc[mt][v] + bias;
    }
  }
}

// ---------------------------------------------------------------------------
// K4: per-channel mean / inv-std over (B,H,W)
// ---------------------------------------------------------------------------
__global__ __launch_bounds__(256)
void k4_stats(const float* __restrict__ z2, float* __restrict__ stats) {
  __shared__ float s1[256], s2[256];
  const int c = blockIdx.x;
  const int tid = threadIdx.x;
  float a = 0.f, q = 0.f;
  for (int bb = 0; bb < B_; ++bb) {
    const float* p = z2 + (bb*COUT + c)*HW;
    for (int i = tid; i < HW; i += 256) { const float v = p[i]; a += v; q += v*v; }
  }
  s1[tid] = a; s2[tid] = q;
  __syncthreads();
  for (int off = 128; off > 0; off >>= 1) {
    if (tid < off) { s1[tid] += s1[tid+off]; s2[tid] += s2[tid+off]; }
    __syncthreads();
  }
  if (tid == 0) {
    const float mean = s1[0] / (float)(B_*HW);
    const float var  = s2[0] / (float)(B_*HW) - mean*mean;
    stats[c*2]   = mean;
    stats[c*2+1] = 1.0f / sqrtf(var + 1e-5f);
  }
}

// ---------------------------------------------------------------------------
// K5: BN + ReLU + 5x5 stride-1 maxpool (fused; z2 is L2-resident)
// ---------------------------------------------------------------------------
__global__ __launch_bounds__(256)
void k5_bn_pool(const float* __restrict__ z2, const float* __restrict__ stats,
                const float* __restrict__ gamma, const float* __restrict__ beta,
                float* __restrict__ out) {
  const int idx = blockIdx.x * 256 + threadIdx.x;
  const int w = idx % W_;
  const int h = (idx / W_) % H_;
  const int c = (idx / HW) % COUT;
  const int b = idx / (COUT*HW);
  const float sc = stats[c*2+1] * gamma[c];
  const float sh = beta[c] - stats[c*2] * sc;
  const float* p = z2 + (b*COUT + c)*HW;
  float best = 0.f;                    // relu'd values are >= 0
  for (int dy = -2; dy <= 2; ++dy) {
    const int y = h + dy; if (y < 0 || y >= H_) continue;
    for (int dx = -2; dx <= 2; ++dx) {
      const int xx = w + dx; if (xx < 0 || xx >= W_) continue;
      const float t = fmaxf(p[y*W_ + xx]*sc + sh, 0.f);
      best = fmaxf(best, t);
    }
  }
  out[idx] = best;
}

// ---------------------------------------------------------------------------
extern "C" void kernel_launch(void* const* d_in, const int* in_sizes, int n_in,
                              void* d_out, int out_size, void* d_ws, size_t ws_size,
                              hipStream_t stream) {
  const float* x      = (const float*)d_in[0];
  const float* conv_w = (const float*)d_in[1];
  const float* conv_b = (const float*)d_in[2];
  const float* gamma  = (const float*)d_in[3];
  const float* beta   = (const float*)d_in[4];
  const float* fcw    = (const float*)d_in[5];
  const float* fcb    = (const float*)d_in[6];

  // workspace layout (bytes):
  //   gidx : 2*1024*49*20*4 =  8,028,160   (offset          0)
  //   z1h  : 2*64*224*224*2 = 12,845,056   (offset  8,028,160)
  //   z2   : 2*128*224*224*4= 51,380,224   (offset 20,873,216)
  //   stats: 128*2*4        =      1,024   (offset 72,253,440)
  char* ws = (char*)d_ws;
  int*      gidx  = (int*)     (ws + 0);
  _Float16* z1h   = (_Float16*)(ws + 8028160);
  float*    z2    = (float*)   (ws + 20873216);
  float*    stats = (float*)   (ws + 72253440);
  float*    out   = (float*)d_out;

  k1_gram_topk<<<B_*NPATCH,        128, 0, stream>>>(x, gidx);
  k2_gather_fc<<<(B_*HW)/8,        256, 0, stream>>>(x, gidx, fcw, fcb, z1h);
  k3_conv_wmma<<<(B_*HW)/64,       256, 0, stream>>>(z1h, conv_w, conv_b, z2);
  k4_stats    <<<COUT,             256, 0, stream>>>(z2, stats);
  k5_bn_pool  <<<(B_*COUT*HW)/256, 256, 0, stream>>>(z2, stats, gamma, beta, out);
}
